// BitLinear_20091857010720
// MI455X (gfx1250) — 8-bit matrix operand order repaired and checked by reading, compiled, not run on hardware
//
#include <hip/hip_runtime.h>
#include <stdint.h>

typedef __attribute__((ext_vector_type(8))) int v8i;

#define EPSQ 1e-8f

__global__ __launch_bounds__(256) void quant_act_kernel(
    const float* __restrict__ x, int8_t* __restrict__ xq,
    float* __restrict__ rowScale, int K)
{
    const int row = blockIdx.x;
    const int tid = threadIdx.x;
    const float* xr = x + (size_t)row * K;

    __shared__ float red[256];
    float amax = 0.0f;
    for (int i = tid; i < K; i += 256)
        amax = fmaxf(amax, fabsf(xr[i]));
    red[tid] = amax;
    __syncthreads();
    #pragma unroll
    for (int s = 128; s > 0; s >>= 1) {
        if (tid < s) red[tid] = fmaxf(red[tid], red[tid + s]);
        __syncthreads();
    }
    amax = fmaxf(red[0], EPSQ);
    const float s = 7.0f / amax;
    if (tid == 0) rowScale[row] = amax / 7.0f;

    int8_t* xqr = xq + (size_t)row * K;
    for (int i = tid; i < K; i += 256) {
        float q = rintf(xr[i] * s);
        q = fminf(fmaxf(q, -8.0f), 7.0f);
        xqr[i] = (int8_t)q;
    }
}

__global__ __launch_bounds__(256) void wabs_partial_kernel(
    const float* __restrict__ w, float* __restrict__ partials, size_t n)
{
    __shared__ float red[256];
    const int tid = threadIdx.x;
    float s = 0.0f;
    const size_t stride = (size_t)256 * gridDim.x;
    for (size_t i = (size_t)blockIdx.x * 256 + tid; i < n; i += stride)
        s += fabsf(w[i]);
    red[tid] = s;
    __syncthreads();
    #pragma unroll
    for (int st = 128; st > 0; st >>= 1) {
        if (tid < st) red[tid] += red[tid + st];
        __syncthreads();
    }
    if (tid == 0) partials[blockIdx.x] = red[0];
}

__global__ __launch_bounds__(256) void wabs_final_kernel(
    const float* __restrict__ partials, float* __restrict__ wscale,
    int np, float invN)
{
    __shared__ float red[256];
    const int tid = threadIdx.x;
    float s = 0.0f;
    for (int i = tid; i < np; i += 256) s += partials[i];
    red[tid] = s;
    __syncthreads();
    #pragma unroll
    for (int st = 128; st > 0; st >>= 1) {
        if (tid < st) red[tid] += red[tid + st];
        __syncthreads();
    }
    if (tid == 0) wscale[0] = fmaxf(red[0] * invN, EPSQ);
}

__global__ __launch_bounds__(256) void ternarize_kernel(
    const float* __restrict__ w, const float* __restrict__ wscale,
    int8_t* __restrict__ wq, size_t n)
{
    const float inv = 1.0f / wscale[0];
    const size_t stride = (size_t)256 * gridDim.x;
    for (size_t i = (size_t)blockIdx.x * 256 + threadIdx.x; i < n; i += stride) {
        float q = rintf(w[i] * inv);
        q = fminf(fmaxf(q, -1.0f), 1.0f);
        wq[i] = (int8_t)q;
    }
}

#define LSTRIDE 80

__global__ __launch_bounds__(256) void bitlinear_gemm_kernel(
    const int8_t* __restrict__ xq, const int8_t* __restrict__ wq,
    const float* __restrict__ rowScale, const float* __restrict__ wscalePtr,
    float* __restrict__ out, int T, int O, int K)
{
    __shared__ __align__(16) int8_t lA[2][128][LSTRIDE];
    __shared__ __align__(16) int8_t lB[2][128][LSTRIDE];

    const int tid   = threadIdx.x;
    const int lane  = tid & 31;
    const int wave  = tid >> 5;
    const int waveM = wave & 3;
    const int waveN = wave >> 2;
    const int half  = lane >> 4;
    const int l16   = lane & 15;

    const int blockM = blockIdx.y * 128;
    const int blockN = blockIdx.x * 128;

    const int r0 = tid >> 2;
    const int cu = (tid & 3) * 16;

    const int8_t* gA0 = xq + (size_t)(blockM + r0) * K + cu;
    const int8_t* gA1 = gA0 + (size_t)64 * K;
    const int8_t* gB0 = wq + (size_t)(blockN + r0) * K + cu;
    const int8_t* gB1 = gB0 + (size_t)64 * K;

    uint32_t dA0[2], dA1[2], dB0[2], dB1[2];
    #pragma unroll
    for (int st = 0; st < 2; ++st) {
        dA0[st] = (uint32_t)(uintptr_t)&lA[st][r0][cu];
        dA1[st] = (uint32_t)(uintptr_t)&lA[st][r0 + 64][cu];
        dB0[st] = (uint32_t)(uintptr_t)&lB[st][r0][cu];
        dB1[st] = (uint32_t)(uintptr_t)&lB[st][r0 + 64][cu];
    }

    auto stage = [&](int st, int k0) {
        asm volatile("global_load_async_to_lds_b128 %0, %1, off"
                     :: "v"(dA0[st]), "v"(gA0 + k0) : "memory");
        asm volatile("global_load_async_to_lds_b128 %0, %1, off"
                     :: "v"(dA1[st]), "v"(gA1 + k0) : "memory");
        asm volatile("global_load_async_to_lds_b128 %0, %1, off"
                     :: "v"(dB0[st]), "v"(gB0 + k0) : "memory");
        asm volatile("global_load_async_to_lds_b128 %0, %1, off"
                     :: "v"(dB1[st]), "v"(gB1 + k0) : "memory");
    };

    v8i zero = {};
    v8i acc[2][4];
    #pragma unroll
    for (int i = 0; i < 2; ++i)
        #pragma unroll
        for (int j = 0; j < 4; ++j)
            acc[i][j] = zero;

    const int NK = K >> 6;
    stage(0, 0);

    for (int kt = 0; kt < NK; ++kt) {
        const int cur = kt & 1;
        asm volatile("s_wait_asynccnt 0x0" ::: "memory");
        __syncthreads();
        if (kt + 1 < NK) stage(cur ^ 1, (kt + 1) << 6);

        v8i aF[2];
        #pragma unroll
        for (int i = 0; i < 2; ++i) {
            const int8_t* p = &lA[cur][waveM * 32 + i * 16 + l16][half * 8];
            int2 c0 = *(const int2*)(p);
            int2 c1 = *(const int2*)(p + 16);
            int2 c2 = *(const int2*)(p + 32);
            int2 c3 = *(const int2*)(p + 48);
            aF[i][0] = c0.x; aF[i][1] = c0.y;
            aF[i][2] = c1.x; aF[i][3] = c1.y;
            aF[i][4] = c2.x; aF[i][5] = c2.y;
            aF[i][6] = c3.x; aF[i][7] = c3.y;
        }
        v8i bF[4];
        #pragma unroll
        for (int j = 0; j < 4; ++j) {
            const int8_t* p = &lB[cur][waveN * 64 + j * 16 + l16][half * 8];
            int2 q0 = *(const int2*)(p);
            int2 q1 = *(const int2*)(p + 16);
            int2 q2 = *(const int2*)(p + 32);
            int2 q3 = *(const int2*)(p + 48);
            bF[j][0] = q0.x; bF[j][1] = q0.y; bF[j][2] = q1.x; bF[j][3] = q1.y;
            bF[j][4] = q2.x; bF[j][5] = q2.y; bF[j][6] = q3.x; bF[j][7] = q3.y;
        }
        #pragma unroll
        for (int i = 0; i < 2; ++i)
            #pragma unroll
            for (int j = 0; j < 4; ++j)
                acc[i][j] = __builtin_amdgcn_wmma_i32_16x16x64_iu8(
                     true, aF[i],  true, bF[j],
                    acc[i][j],  false,  false);

        __syncthreads();
    }

    const float wscale = *wscalePtr;
    #pragma unroll
    for (int i = 0; i < 2; ++i) {
        const int mBase = blockM + waveM * 32 + i * 16 + half * 8;
        float rs[8];
        #pragma unroll
        for (int r = 0; r < 8; ++r)
            rs[r] = rowScale[mBase + r] * wscale;
        #pragma unroll
        for (int j = 0; j < 4; ++j) {
            const int gn = blockN + waveN * 64 + j * 16 + l16;
            float* op = out + (size_t)mBase * O + gn;
            #pragma unroll
            for (int r = 0; r < 8; ++r)
                op[(size_t)r * O] = (float)acc[i][j][r] * rs[r];
        }
    }
}

extern "C" void kernel_launch(void* const* d_in, const int* in_sizes, int n_in,
                              void* d_out, int out_size, void* d_ws, size_t ws_size,
                              hipStream_t stream)
{
    const float* x = (const float*)d_in[0];
    const float* w = (const float*)d_in[1];
    float* out = (float*)d_out;

    const int K = 4096;
    const int O = 4096;
    const int T = in_sizes[0] / K;
    const size_t nW = (size_t)O * K;

    uint8_t* ws = (uint8_t*)d_ws;
    int8_t* xq        = (int8_t*)ws;
    int8_t* wqi       = (int8_t*)(ws + (size_t)T * K);
    float*  rowScale  = (float*)(ws + (size_t)T * K + nW);
    float*  partials  = rowScale + T;
    float*  wscale    = partials + 1024;

    quant_act_kernel<<<T, 256, 0, stream>>>(x, xq, rowScale, K);
    wabs_partial_kernel<<<1024, 256, 0, stream>>>(w, partials, nW);
    wabs_final_kernel<<<1, 256, 0, stream>>>(partials, wscale, 1024,
                                             1.0f / (float)nW);
    ternarize_kernel<<<4096, 256, 0, stream>>>(w, wscale, wqi, nW);

    dim3 grid(O / 128, T / 128);
    bitlinear_gemm_kernel<<<grid, 256, 0, stream>>>(xq, wqi, rowScale, wscale,
                                                    out, T, O, K);
}
